// LSTM_24910810317337
// MI455X (gfx1250) — matrix operand order repaired and checked by reading, not compiled, not run on hardware
//
#include <hip/hip_runtime.h>
#include <hip/hip_bf16.h>

// ---------------------------------------------------------------------------
// LSTM (S=1024, B=64, D=H=1024) for gfx1250 (MI455X, wave32, WMMA).
//
// Phase 1: prep      — pack 4 input / 4 hidden weight mats to bf16, fuse biases
// Phase 2: xgemm     — preact_x[S*B, 4H] = X[S*B, D] @ Wi^T  (bf16 WMMA, f32 acc)
// Phase 3: 1024x step — gates = preact_x[s] + h @ Wh^T + bias; LSTM cell update
// Phase 4: finalize  — emit hT, cT tail of d_out
// ---------------------------------------------------------------------------

#define LSTM_S 1024
#define LSTM_B 64
#define LSTM_D 1024
#define LSTM_H 1024
#define LSTM_4H 4096

typedef __bf16 bf16_t;
typedef __attribute__((ext_vector_type(16))) __bf16 v16bf;
typedef __attribute__((ext_vector_type(8)))  __bf16 v8bf;
typedef __attribute__((ext_vector_type(8)))  float  v8f;

// ---- WMMA fragment helpers (layouts per CDNA5 ISA 7.12.2, wave32) ----------

// A fragment: 16x32 bf16, row-major source with row stride `ld` elements.
// lane m (0-15): K = {0..7, 16..23}; lane m+16: K = {8..15, 24..31}.
__device__ __forceinline__ v16bf load_frag_a(const bf16_t* p, int ld, int lane) {
    int m   = lane & 15;
    int sel = (lane >> 4) * 8;
    const bf16_t* r = p + (size_t)m * ld + sel;
    union { v16bf v; v8bf h[2]; } u;
    u.h[0] = *(const v8bf*)(r);        // K 0..7   (+8 for upper half-wave)
    u.h[1] = *(const v8bf*)(r + 16);   // K 16..23 (+8 for upper half-wave)
    return u.v;
}

// B fragment: 32x16 bf16 (KxN). Source is N-major [N][K] (i.e. W rows), so
// lane n (0-15): K = {0..7, 16..23}; lane n+16: K = {8..15, 24..31}.
__device__ __forceinline__ v16bf load_frag_b(const bf16_t* p, int ld, int lane) {
    int n   = lane & 15;
    int sel = (lane >> 4) * 8;
    const bf16_t* r = p + (size_t)n * ld + sel;
    union { v16bf v; v8bf h[2]; } u;
    u.h[0] = *(const v8bf*)(r);
    u.h[1] = *(const v8bf*)(r + 16);
    return u.v;
}

__device__ __forceinline__ v8f wmma_bf16(v16bf a, v16bf b, v8f c) {
    return __builtin_amdgcn_wmma_f32_16x16x32_bf16(
        /*neg_a=*/false, a, /*neg_b=*/false, b,
        /*c_mod=*/(short)0, c, /*reuse_a=*/false, /*reuse_b=*/false);
}

// C/D tile (16x16 f32): row = vgpr + (lane>>4)*8, col = lane&15.
__device__ __forceinline__ void store_tile_bf16(bf16_t* p, int ld, int lane, v8f acc) {
    int col = lane & 15;
    int r0  = (lane >> 4) * 8;
#pragma unroll
    for (int r = 0; r < 8; ++r)
        p[(size_t)(r0 + r) * ld + col] = (bf16_t)acc[r];
}

__device__ __forceinline__ float sigm_f(float x) { return 1.0f / (1.0f + __expf(-x)); }
__device__ __forceinline__ float tanh_f(float x) { return 1.0f - 2.0f / (__expf(2.0f * x) + 1.0f); }

// ---------------------------------------------------------------------------
// Phase 1: pack weights -> bf16 [4][H][K], fuse biases. grid(4096,4) x 256
// ---------------------------------------------------------------------------
__global__ void lstm_prep(const float* iw0, const float* iw1, const float* iw2, const float* iw3,
                          const float* hw0, const float* hw1, const float* hw2, const float* hw3,
                          const float* ib0, const float* ib1, const float* ib2, const float* ib3,
                          const float* hb0, const float* hb1, const float* hb2, const float* hb3,
                          bf16_t* __restrict__ Wi, bf16_t* __restrict__ Wh,
                          float* __restrict__ bias) {
    int g = blockIdx.y;
    size_t r = (size_t)blockIdx.x * 256 + threadIdx.x;          // 0 .. H*D-1
    const float* iw = (g == 0) ? iw0 : (g == 1) ? iw1 : (g == 2) ? iw2 : iw3;
    const float* hw = (g == 0) ? hw0 : (g == 1) ? hw1 : (g == 2) ? hw2 : hw3;
    size_t o = (size_t)g * (LSTM_H * (size_t)LSTM_D) + r;
    Wi[o] = (bf16_t)iw[r];
    Wh[o] = (bf16_t)hw[r];
    if (r < LSTM_H) {
        const float* ib = (g == 0) ? ib0 : (g == 1) ? ib1 : (g == 2) ? ib2 : ib3;
        const float* hb = (g == 0) ? hb0 : (g == 1) ? hb1 : (g == 2) ? hb2 : hb3;
        bias[g * LSTM_H + r] = ib[r] + hb[r];
    }
}

// ---------------------------------------------------------------------------
// Zero initial state h0 (bf16) and c (f32). grid 256 x 256 (B*H = 65536)
// ---------------------------------------------------------------------------
__global__ void lstm_init(bf16_t* __restrict__ h0, float* __restrict__ c) {
    int i = blockIdx.x * 256 + threadIdx.x;
    h0[i] = (bf16_t)0.0f;
    c[i]  = 0.0f;
}

// ---------------------------------------------------------------------------
// Phase 2: preact[M=65536][N=4096] = X[M][1024] @ Wi^T (Wi is [4096][1024]).
// Block tile 64(M) x 128(N), 8 waves, each wave 32x32 (2x2 WMMA tiles).
// A is converted fp32->bf16 on the fly into LDS. grid(32, 1024) x 256.
// ---------------------------------------------------------------------------
__global__ void lstm_xgemm(const float* __restrict__ X,
                           const bf16_t* __restrict__ Wi,
                           bf16_t* __restrict__ P) {
    __shared__ bf16_t As[64][40];    // 64x32 tile, stride 40 keeps 16B align + spreads banks
    __shared__ bf16_t Bs[128][40];

    const int tid  = threadIdx.x;
    const int lane = tid & 31;
    const int w    = tid >> 5;          // 0..7
    const int wm   = w >> 2;            // 0..1  -> m sub-block *32
    const int wn   = w & 3;             // 0..3  -> n sub-block *32
    const int nb   = blockIdx.x;        // 0..31   n0 = nb*128
    const int mb   = blockIdx.y;        // 0..1023 m0 = mb*64

    const int ar = tid >> 2;            // 0..63  A row loaded by this thread
    const int ac = (tid & 3) * 8;       // 0/8/16/24
    const int br = tid >> 1;            // 0..127 B row
    const int bc = (tid & 1) * 16;      // 0/16

    const float*  Arow = X  + (size_t)(mb * 64 + ar) * LSTM_D + ac;
    const bf16_t* Brow = Wi + (size_t)(nb * 128 + br) * LSTM_D + bc;

    v8f acc00 = {}, acc01 = {}, acc10 = {}, acc11 = {};

    for (int kc = 0; kc < LSTM_D; kc += 32) {
        __syncthreads();
        // Cooperative A load: fp32 -> bf16 into LDS (8 elems / thread).
        {
            const float* src = Arow + kc;
            float4 f0 = *(const float4*)(src);
            float4 f1 = *(const float4*)(src + 4);
            v8bf pack = { (bf16_t)f0.x, (bf16_t)f0.y, (bf16_t)f0.z, (bf16_t)f0.w,
                          (bf16_t)f1.x, (bf16_t)f1.y, (bf16_t)f1.z, (bf16_t)f1.w };
            *(v8bf*)&As[ar][ac] = pack;
        }
        // Cooperative B load: 16 bf16 / thread.
        {
            const bf16_t* src = Brow + kc;
            v8bf b0 = *(const v8bf*)(src);
            v8bf b1 = *(const v8bf*)(src + 8);
            *(v8bf*)&Bs[br][bc]     = b0;
            *(v8bf*)&Bs[br][bc + 8] = b1;
        }
        __syncthreads();

        v16bf a0 = load_frag_a(&As[wm * 32][0],      40, lane);
        v16bf a1 = load_frag_a(&As[wm * 32 + 16][0], 40, lane);
        v16bf b0 = load_frag_b(&Bs[wn * 32][0],      40, lane);
        v16bf b1 = load_frag_b(&Bs[wn * 32 + 16][0], 40, lane);

        acc00 = wmma_bf16(a0, b0, acc00);
        acc01 = wmma_bf16(a0, b1, acc01);
        acc10 = wmma_bf16(a1, b0, acc10);
        acc11 = wmma_bf16(a1, b1, acc11);
    }

    const size_t m0 = (size_t)mb * 64 + wm * 32;
    const size_t n0 = (size_t)nb * 128 + wn * 32;
    store_tile_bf16(P + (m0)      * LSTM_4H + n0,      LSTM_4H, lane, acc00);
    store_tile_bf16(P + (m0)      * LSTM_4H + n0 + 16, LSTM_4H, lane, acc01);
    store_tile_bf16(P + (m0 + 16) * LSTM_4H + n0,      LSTM_4H, lane, acc10);
    store_tile_bf16(P + (m0 + 16) * LSTM_4H + n0 + 16, LSTM_4H, lane, acc11);
}

// ---------------------------------------------------------------------------
// Phase 3: one recurrent step. grid = 64 blocks (j-tiles of 16), 256 threads.
// Block computes all 4 gates for its j-slice: 8 waves = 4 gates x 2 M-halves,
// each wave 2 accumulators (M 32 rows). Wh (8 MB bf16) stays L2-resident.
// ---------------------------------------------------------------------------
__global__ void lstm_step(const bf16_t* __restrict__ preact_s, // [64][4096]
                          const bf16_t* __restrict__ h_in,     // [64][1024] bf16
                          const bf16_t* __restrict__ Wh,       // [4096][1024]
                          const float*  __restrict__ bias,     // [4096]
                          float*        __restrict__ c,        // [64][1024]
                          bf16_t*       __restrict__ h_out,    // [64][1024]
                          float*        __restrict__ y) {      // [64][1024]
    __shared__ float gbuf[4][64][16];   // gate preacts (recurrent part), 16 KB

    const int tid  = threadIdx.x;
    const int lane = tid & 31;
    const int w    = tid >> 5;
    const int jt   = blockIdx.x;        // 0..63
    const int g    = w >> 1;            // gate 0..3 (I,F,G,O)
    const int mh   = (w & 1) * 32;      // M half

    const bf16_t* Brow = Wh + ((size_t)g * LSTM_H + (size_t)jt * 16) * LSTM_H;
    const bf16_t* A0   = h_in + (size_t)mh * LSTM_H;
    const bf16_t* A1   = h_in + (size_t)(mh + 16) * LSTM_H;

    v8f acc0 = {}, acc1 = {};
    for (int kc = 0; kc < LSTM_H; kc += 32) {
        __builtin_prefetch(Brow + kc + 64, 0, 1);   // global_prefetch next Wh chunk
        v16bf bfrag = load_frag_b(Brow + kc, LSTM_H, lane);
        v16bf a0    = load_frag_a(A0 + kc,   LSTM_H, lane);
        v16bf a1    = load_frag_a(A1 + kc,   LSTM_H, lane);
        acc0 = wmma_bf16(a0, bfrag, acc0);
        acc1 = wmma_bf16(a1, bfrag, acc1);
    }

    // Park WMMA tiles in LDS for cross-gate fusion.
    {
        int col = lane & 15;
        int r0  = (lane >> 4) * 8;
#pragma unroll
        for (int r = 0; r < 8; ++r) {
            gbuf[g][mh + r0 + r][col]      = acc0[r];
            gbuf[g][mh + 16 + r0 + r][col] = acc1[r];
        }
    }
    __syncthreads();

    // Elementwise LSTM cell update: 64x16 = 1024 elems, 4 per thread.
#pragma unroll
    for (int it = 0; it < 4; ++it) {
        int i   = tid + it * 256;
        int b   = i >> 4;
        int col = i & 15;
        int j   = jt * 16 + col;
        size_t pb = (size_t)b * LSTM_4H;

        float pI = gbuf[0][b][col] + (float)preact_s[pb + 0 * LSTM_H + j] + bias[0 * LSTM_H + j];
        float pF = gbuf[1][b][col] + (float)preact_s[pb + 1 * LSTM_H + j] + bias[1 * LSTM_H + j];
        float pG = gbuf[2][b][col] + (float)preact_s[pb + 2 * LSTM_H + j] + bias[2 * LSTM_H + j];
        float pO = gbuf[3][b][col] + (float)preact_s[pb + 3 * LSTM_H + j] + bias[3 * LSTM_H + j];

        float I = sigm_f(pI);
        float F = sigm_f(pF);
        float G = tanh_f(pG);
        float O = sigm_f(pO);

        size_t o  = (size_t)b * LSTM_H + j;
        float  cn = F * c[o] + I * G;
        c[o] = cn;
        float yv = O * tanh_f(cn);
        y[o]     = yv;
        h_out[o] = (bf16_t)yv;
    }
}

// ---------------------------------------------------------------------------
// Phase 4: hT = Y[S-1] (fp32, exact), cT = c. grid 256 x 256.
// ---------------------------------------------------------------------------
__global__ void lstm_final(const float* __restrict__ ylast,
                           const float* __restrict__ c,
                           float* __restrict__ tail) {
    int i = blockIdx.x * 256 + threadIdx.x;
    tail[i]                     = ylast[i];
    tail[LSTM_B * LSTM_H + i]   = c[i];
}

// ---------------------------------------------------------------------------
extern "C" void kernel_launch(void* const* d_in, const int* in_sizes, int n_in,
                              void* d_out, int out_size, void* d_ws, size_t ws_size,
                              hipStream_t stream) {
    (void)in_sizes; (void)n_in; (void)out_size; (void)ws_size;

    const float* Xt = (const float*)d_in[0];
    // Gate order I,F,G,O; inputs interleaved (iiW,hiW,ifW,hfW,...).
    const float* iw[4] = { (const float*)d_in[1], (const float*)d_in[3],
                           (const float*)d_in[5], (const float*)d_in[7] };
    const float* hw[4] = { (const float*)d_in[2], (const float*)d_in[4],
                           (const float*)d_in[6], (const float*)d_in[8] };
    const float* ib[4] = { (const float*)d_in[9],  (const float*)d_in[11],
                           (const float*)d_in[13], (const float*)d_in[15] };
    const float* hb[4] = { (const float*)d_in[10], (const float*)d_in[12],
                           (const float*)d_in[14], (const float*)d_in[16] };

    char* ws = (char*)d_ws;
    size_t off = 0;
    auto alloc = [&](size_t bytes) -> void* {
        void* p = ws + off;
        off = (off + bytes + 255) & ~(size_t)255;
        return p;
    };

    bf16_t* Wi     = (bf16_t*)alloc((size_t)LSTM_4H * LSTM_D * sizeof(bf16_t));  // 8 MiB
    bf16_t* Wh     = (bf16_t*)alloc((size_t)LSTM_4H * LSTM_H * sizeof(bf16_t));  // 8 MiB
    float*  bias   = (float*) alloc((size_t)LSTM_4H * sizeof(float));
    bf16_t* hbuf   = (bf16_t*)alloc((size_t)2 * LSTM_B * LSTM_H * sizeof(bf16_t));
    float*  cbuf   = (float*) alloc((size_t)LSTM_B * LSTM_H * sizeof(float));
    bf16_t* preact = (bf16_t*)alloc((size_t)LSTM_S * LSTM_B * LSTM_4H * sizeof(bf16_t)); // 512 MiB

    // Phase 1: weight pack + bias fuse.
    lstm_prep<<<dim3(LSTM_H * LSTM_D / 256, 4), 256, 0, stream>>>(
        iw[0], iw[1], iw[2], iw[3], hw[0], hw[1], hw[2], hw[3],
        ib[0], ib[1], ib[2], ib[3], hb[0], hb[1], hb[2], hb[3],
        Wi, Wh, bias);

    // Zero h0 / c0.
    lstm_init<<<dim3(LSTM_B * LSTM_H / 256), 256, 0, stream>>>(hbuf, cbuf);

    // Phase 2: batched input projection for all timesteps.
    lstm_xgemm<<<dim3(LSTM_4H / 128, LSTM_S * LSTM_B / 64), 256, 0, stream>>>(
        Xt, Wi, preact);

    // Phase 3: sequential recurrence (stream order provides the dependency).
    float* Y = (float*)d_out;
    for (int s = 0; s < LSTM_S; ++s) {
        const bf16_t* hin  = hbuf + (size_t)(s & 1) * LSTM_B * LSTM_H;
        bf16_t*       hout = hbuf + (size_t)((s + 1) & 1) * LSTM_B * LSTM_H;
        lstm_step<<<dim3(LSTM_H / 16), 256, 0, stream>>>(
            preact + (size_t)s * LSTM_B * LSTM_4H,
            hin, Wh, bias, cbuf, hout,
            Y + (size_t)s * LSTM_B * LSTM_H);
    }

    // Phase 4: hT (exact, from fp32 Y) and cT.
    lstm_final<<<dim3(LSTM_B * LSTM_H / 256), 256, 0, stream>>>(
        Y + (size_t)(LSTM_S - 1) * LSTM_B * LSTM_H, cbuf,
        Y + (size_t)LSTM_S * LSTM_B * LSTM_H);
}
